// RNN_24206435680652
// MI455X (gfx1250) — matrix operand order repaired and checked by reading, not compiled, not run on hardware
//
#include <hip/hip_runtime.h>
#include <hip/hip_bf16.h>
#include <math.h>

// Problem sizes (fixed by the reference)
#define B_ 64
#define S_ 1024
#define I_ 512
#define H_ 1024
#define O_ 256

typedef __attribute__((ext_vector_type(16))) __bf16 v16bf;
typedef __attribute__((ext_vector_type(8)))  __bf16 v8bf;
typedef __attribute__((ext_vector_type(4)))  __bf16 v4bf;
typedef __attribute__((ext_vector_type(8)))  float  v8f;
typedef __attribute__((ext_vector_type(4)))  float  v4f;

// ---------- bf16 helpers: native conversions (backend picks v_cvt_*) ----------
__device__ __forceinline__ __bf16 f2bf(float f) { return (__bf16)f; }
__device__ __forceinline__ float  bf2f(__bf16 h) { return (float)h; }

// ---------- WMMA fragment loaders (wave32, 16x16x32 bf16) ----------
// A (16x32, MxK): lane&15 = row; low lanes hold K = {0..7,16..23}, high lanes
// hold K = {8..15,24..31} (ISA 7.12.2, 16-bit A 16x32 table).
// Two aligned 16B vector loads per lane.
__device__ __forceinline__ v16bf load_a_bf16(const __bf16* row0, long row_stride, int lane) {
  int r = lane & 15;
  int koff = (lane >> 4) << 3;
  const __bf16* p = row0 + (long)r * row_stride + koff;
  v8bf lo = *(const v8bf*)p;
  v8bf hi = *(const v8bf*)(p + 16);
  return __builtin_shufflevector(lo, hi, 0,1,2,3,4,5,6,7,8,9,10,11,12,13,14,15);
}
// B (32x16, KxN): lane&15 = column N; low lanes K={0..7,16..23}, high lanes K={8..15,24..31}.
// B[k][n] = W[n][k] for row-major W -> two aligned 16B loads per lane.
__device__ __forceinline__ v16bf load_b_bf16(const __bf16* wrow0, long row_stride, int lane) {
  int c  = lane & 15;
  int kb = (lane >> 4) << 3;
  const __bf16* p = wrow0 + (long)c * row_stride + kb;
  v8bf lo = *(const v8bf*)p;
  v8bf hi = *(const v8bf*)(p + 16);
  return __builtin_shufflevector(lo, hi, 0,1,2,3,4,5,6,7,8,9,10,11,12,13,14,15);
}

#define WMMA_BF16(A, Bf, C) \
  __builtin_amdgcn_wmma_f32_16x16x32_bf16(false, (A), false, (Bf), (short)0, (C), false, false)

// ================================================================
// Kernel 0: bulk f32 -> bf16 conversion (vectorized x4, grid-strided)
// ================================================================
__global__ void cvt_bf16_kernel(const float* __restrict__ src, __bf16* __restrict__ dst, long n4) {
  long i = (long)blockIdx.x * blockDim.x + threadIdx.x;
  const long stride = (long)gridDim.x * blockDim.x;
  for (; i < n4; i += stride) {
    v4f v = ((const v4f*)src)[i];
    ((v4bf*)dst)[i] = __builtin_convertvector(v, v4bf);
  }
}

// ================================================================
// Kernel 1: xp[s,b,h] = bf16( x[b,s,:] . Wx[h,:] + bias[h] )
// All-bf16 operands. Grid (4096 M-tiles, 16 N-groups), block 128 = 4 waves.
// ================================================================
__global__ void xproj_kernel(const __bf16* __restrict__ xb, const __bf16* __restrict__ Wxb,
                             const float* __restrict__ bias, __bf16* __restrict__ xp) {
  const int lane = threadIdx.x & 31;
  const int wave = threadIdx.x >> 5;
  const int mt = blockIdx.x;            // 0..4095 : m-tile over (s, b0)
  const int s  = mt >> 2;
  const int b0 = (mt & 3) << 4;
  const int j0 = ((blockIdx.y << 2) + wave) << 4;   // H column tile

  const __bf16* arow = xb  + ((size_t)b0 * S_ + s) * I_;  // A row stride = S_*I_
  const __bf16* brow = Wxb + (size_t)j0 * I_;

  v8f acc0 = {0.f,0.f,0.f,0.f,0.f,0.f,0.f,0.f};
  v8f acc1 = acc0;
  for (int kk = 0; kk < I_; kk += 64) {
    v16bf a0 = load_a_bf16(arow + kk,      (long)S_ * I_, lane);
    v16bf a1 = load_a_bf16(arow + kk + 32, (long)S_ * I_, lane);
    v16bf w0 = load_b_bf16(brow + kk,      I_, lane);
    v16bf w1 = load_b_bf16(brow + kk + 32, I_, lane);
    acc0 = WMMA_BF16(a0, w0, acc0);
    acc1 = WMMA_BF16(a1, w1, acc1);
  }
  v8f acc = acc0 + acc1;
  const int n    = lane & 15;
  const int mrow = (lane >> 4) << 3;
  const float bv = bias[j0 + n];
#pragma unroll
  for (int v = 0; v < 8; ++v) {
    int bidx = b0 + mrow + v;
    xp[((size_t)s * B_ + bidx) * H_ + j0 + n] = f2bf(acc[v] + bv);
  }
}

// ================================================================
// Kernel 2: persistent cooperative scan.
// 64 workgroups x 128 threads (4 waves). Block owns 16 H-columns (Wh strip
// cached in LDS as bf16, 32 KB). Wave owns 16 batch rows. h ping-pongs in a
// global bf16 buffer (lives in L2); one device-scope atomic barrier per step.
// ================================================================
__global__ void rnn_scan_kernel(const float* __restrict__ Wh,
                                const __bf16* __restrict__ xp,
                                float* __restrict__ all_outs,
                                __bf16* __restrict__ hbuf,     // [2][B_][H_]
                                unsigned* __restrict__ barrier_cnt,
                                int nwg) {
  __shared__ __bf16 lds_wh[16 * H_];   // Wh[j0..j0+15][0..H_-1], 32 KB

  const int lane = threadIdx.x & 31;
  const int wave = threadIdx.x >> 5;
  const int j0 = blockIdx.x << 4;      // this block's 16 output columns
  const int b0 = wave << 4;            // this wave's 16 batch rows

  // one-time Wh strip load: fp32 global -> bf16 LDS
  for (int idx = threadIdx.x; idx < 16 * H_; idx += blockDim.x) {
    int c = idx >> 10;                 // H_ == 1024
    int k = idx & (H_ - 1);
    lds_wh[idx] = f2bf(Wh[(size_t)(j0 + c) * H_ + k]);
  }
  __syncthreads();

  const int n    = lane & 15;
  const int mrow = (lane >> 4) << 3;

  for (int t = 0; t < S_; ++t) {
    const __bf16* hread  = hbuf + (size_t)(t & 1) * B_ * H_;
    __bf16*       hwrite = hbuf + (size_t)((t + 1) & 1) * B_ * H_;

    // 4 independent accumulator chains to hide WMMA latency at 1 wave/SIMD
    v8f acc0 = {0.f,0.f,0.f,0.f,0.f,0.f,0.f,0.f};
    v8f acc1 = acc0, acc2 = acc0, acc3 = acc0;
    const __bf16* arow = hread + (size_t)b0 * H_;
    for (int kk = 0; kk < H_; kk += 128) {
      v16bf a0 = load_a_bf16(arow + kk,       H_, lane);
      v16bf a1 = load_a_bf16(arow + kk + 32,  H_, lane);
      v16bf a2 = load_a_bf16(arow + kk + 64,  H_, lane);
      v16bf a3 = load_a_bf16(arow + kk + 96,  H_, lane);
      v16bf w0 = load_b_bf16(&lds_wh[kk],       H_, lane);
      v16bf w1 = load_b_bf16(&lds_wh[kk + 32],  H_, lane);
      v16bf w2 = load_b_bf16(&lds_wh[kk + 64],  H_, lane);
      v16bf w3 = load_b_bf16(&lds_wh[kk + 96],  H_, lane);
      acc0 = WMMA_BF16(a0, w0, acc0);
      acc1 = WMMA_BF16(a1, w1, acc1);
      acc2 = WMMA_BF16(a2, w2, acc2);
      acc3 = WMMA_BF16(a3, w3, acc3);
    }
    v8f acc = acc0 + acc1 + acc2 + acc3;

    // epilogue: h_new = tanh(acc + xp[t]); write all_outs (f32) + h (bf16)
#pragma unroll
    for (int v = 0; v < 8; ++v) {
      int bb = b0 + mrow + v;
      size_t oidx = ((size_t)t * B_ + bb) * H_ + j0 + n;
      float val = tanhf(acc[v] + bf2f(xp[oidx]));
      all_outs[oidx] = val;
      hwrite[(size_t)bb * H_ + j0 + n] = f2bf(val);
    }

    // device-wide barrier: monotonically increasing counter, target = nwg*(t+1)
    __threadfence();
    __syncthreads();
    if (threadIdx.x == 0) {
      __hip_atomic_fetch_add(barrier_cnt, 1u, __ATOMIC_ACQ_REL, __HIP_MEMORY_SCOPE_AGENT);
      const unsigned target = (unsigned)nwg * (unsigned)(t + 1);
      while (__hip_atomic_load(barrier_cnt, __ATOMIC_ACQUIRE, __HIP_MEMORY_SCOPE_AGENT) < target)
        __builtin_amdgcn_s_sleep(2);
    }
    __syncthreads();
    __threadfence();
  }
}

// ================================================================
// Kernel 3: y[b,o] = h_last[b,:] . Wout[o,:] + bout[o]   (64x256, K=1024)
// 16 blocks x 4 waves, wave-per-16x16-tile, all-bf16 operands.
// ================================================================
__global__ void readout_kernel(const __bf16* __restrict__ hlast,
                               const __bf16* __restrict__ Woutb,
                               const float* __restrict__ bout,
                               float* __restrict__ y) {
  const int lane = threadIdx.x & 31;
  const int wave = threadIdx.x >> 5;
  const int id = blockIdx.x * 4 + wave;   // 0..63
  const int b0 = (id & 3) << 4;
  const int o0 = (id >> 2) << 4;

  v8f acc0 = {0.f,0.f,0.f,0.f,0.f,0.f,0.f,0.f};
  v8f acc1 = acc0;
  for (int kk = 0; kk < H_; kk += 64) {
    v16bf a0 = load_a_bf16(hlast + (size_t)b0 * H_ + kk,      H_, lane);
    v16bf a1 = load_a_bf16(hlast + (size_t)b0 * H_ + kk + 32, H_, lane);
    v16bf w0 = load_b_bf16(Woutb + (size_t)o0 * H_ + kk,      H_, lane);
    v16bf w1 = load_b_bf16(Woutb + (size_t)o0 * H_ + kk + 32, H_, lane);
    acc0 = WMMA_BF16(a0, w0, acc0);
    acc1 = WMMA_BF16(a1, w1, acc1);
  }
  v8f acc = acc0 + acc1;
  const int n    = lane & 15;
  const int mrow = (lane >> 4) << 3;
#pragma unroll
  for (int v = 0; v < 8; ++v)
    y[(size_t)(b0 + mrow + v) * O_ + o0 + n] = acc[v] + bout[o0 + n];
}

// ================================================================
extern "C" void kernel_launch(void* const* d_in, const int* in_sizes, int n_in,
                              void* d_out, int out_size, void* d_ws, size_t ws_size,
                              hipStream_t stream) {
  const float* x    = (const float*)d_in[0];
  const float* Wx   = (const float*)d_in[1];
  const float* Wh   = (const float*)d_in[2];
  const float* bias = (const float*)d_in[3];
  const float* Wout = (const float*)d_in[4];
  const float* bout = (const float*)d_in[5];

  float* y        = (float*)d_out;                        // [B, O]
  float* all_outs = (float*)d_out + (size_t)B_ * O_;      // [S, B, H]

  // workspace layout (bf16 scratch):
  //   xp      [S,B,H]   128 MiB
  //   x_bf16  [B,S,I]    64 MiB
  //   Wx_bf16 [H,I]       1 MiB
  //   Wout_bf16 [O,H]   0.5 MiB
  //   hbuf    [2,B,H]  256 KiB (ping-pong)
  //   barrier counter
  char* ws = (char*)d_ws;
  const size_t xp_elems   = (size_t)S_ * B_ * H_;
  const size_t x_elems    = (size_t)B_ * S_ * I_;
  const size_t wx_elems   = (size_t)H_ * I_;
  const size_t wout_elems = (size_t)O_ * H_;
  const size_t h_elems    = (size_t)2 * B_ * H_;

  __bf16* xp    = (__bf16*)ws;
  __bf16* xb    = xp    + xp_elems;
  __bf16* Wxb   = xb    + x_elems;
  __bf16* Woutb = Wxb   + wx_elems;
  __bf16* hbuf  = Woutb + wout_elems;
  unsigned* cnt = (unsigned*)(hbuf + h_elems);

  // zero h0 (both ping-pong buffers; bf16 zero is 0x0000) and barrier counter
  hipMemsetAsync(hbuf, 0, h_elems * sizeof(__bf16) + 64, stream);

  dim3 blk(128);
  // f32 -> bf16 pre-conversion passes (x4 vectorized)
  cvt_bf16_kernel<<<dim3(4096), dim3(256), 0, stream>>>(x,    xb,    (long)(x_elems / 4));
  cvt_bf16_kernel<<<dim3(512),  dim3(256), 0, stream>>>(Wx,   Wxb,   (long)(wx_elems / 4));
  cvt_bf16_kernel<<<dim3(256),  dim3(256), 0, stream>>>(Wout, Woutb, (long)(wout_elems / 4));

  xproj_kernel   <<<dim3(4096, 16), blk, 0, stream>>>(xb, Wxb, bias, xp);
  rnn_scan_kernel<<<dim3(64),       blk, 0, stream>>>(Wh, xp, all_outs, hbuf, cnt, 64);
  // after 1024 steps, h_last lives in ping-pong buffer 0
  readout_kernel <<<dim3(16),       blk, 0, stream>>>(hbuf, Woutb, bout, y);
}
